// EncoderBlock_71476845740162
// MI455X (gfx1250) — matrix operand order repaired and checked by reading, not compiled, not run on hardware
//
#include <hip/hip_runtime.h>

// ---------------------------------------------------------------------------
// Transformer encoder block for MI455X (gfx1250), wave32 + WMMA bf16.
// B=4, S=2048, D=1024, H=16, DK=64, DFF=4096.  ~275 GFLOP -> matrix-core
// bound: every GEMM runs on v_wmma_f32_16x16x32_bf16.  Weights are pre-packed
// into WMMA B-fragment order (no LDS / no packing VALU in the GEMM hot loop);
// attention V tiles are staged to LDS by the Tensor Data Mover (double
// buffered, overlapped with compute) when the builtin is available.
// ---------------------------------------------------------------------------

#define D_MODEL 1024
#define HEADS   16
#define DK      64
#define DFF     4096
#define BATCH   4
#define SEQ     2048
#define ROWS    (BATCH * SEQ)   // 8192

typedef __attribute__((ext_vector_type(16))) __bf16        v16bf;
typedef __attribute__((ext_vector_type(8)))  float         v8f;
typedef __attribute__((ext_vector_type(8)))  unsigned int  u32x8;
typedef __attribute__((ext_vector_type(4)))  unsigned int  u32x4;
typedef __attribute__((ext_vector_type(8)))  int           i32x8;
typedef __attribute__((ext_vector_type(4)))  int           i32x4;

#if defined(__gfx1250__) && __has_builtin(__builtin_amdgcn_tensor_load_to_lds) && \
    __has_builtin(__builtin_amdgcn_s_wait_tensorcnt)
#define USE_TDM 1
#else
#define USE_TDM 0
#endif

// float -> bf16, round to nearest even
__device__ __forceinline__ unsigned short f2bf(float f) {
    unsigned int u = __builtin_bit_cast(unsigned int, f);
    unsigned int r = u + 0x7FFFu + ((u >> 16) & 1u);
    return (unsigned short)(r >> 16);
}

// CDNA5 16-bit A-matrix (16x32): lane = row (both halves), lanes 0-15 cover
// K {0..7,16..23}, lanes 16-31 cover K {8..15,24..31}; VGPR j packs (k,k+1).
__device__ __forceinline__ int a_kpair(int j, int half) {
    return ((j < 4) ? (2 * j) : (16 + 2 * (j - 4))) + 8 * half;
}

__device__ __forceinline__ v8f wmma_bf16(u32x8 a, u32x8 b, v8f c) {
    return __builtin_amdgcn_wmma_f32_16x16x32_bf16(
        false, __builtin_bit_cast(v16bf, a),
        false, __builtin_bit_cast(v16bf, b),
        (short)0, c, false, false);
}

// B-fragment from a row-major run, p at 8*half (B layout: lane = col,
// k = 8*half + 2j for j<4, 16 + 8*half + 2(j-4) for j>=4): two b128 loads.
__device__ __forceinline__ u32x8 load_frag_contig(const unsigned short* p) {
    const uint4 lo = *(const uint4*)p;
    const uint4 hi = *(const uint4*)(p + 16);
    u32x8 b;
    b[0] = lo.x; b[1] = lo.y; b[2] = lo.z; b[3] = lo.w;
    b[4] = hi.x; b[5] = hi.y; b[6] = hi.z; b[7] = hi.w;
    return b;
}

// scale both bf16 halves of a packed word by 0.125 (exact: exponent shift)
__device__ __forceinline__ unsigned scale_pk(unsigned w) {
    float lo = __builtin_bit_cast(float, (w & 0xffffu) << 16) * 0.125f;
    float hi = __builtin_bit_cast(float, w & 0xffff0000u) * 0.125f;
    return (unsigned)f2bf(lo) | ((unsigned)f2bf(hi) << 16);
}

#if USE_TDM
// Tensor DMA: 2D tile (tile_d0 x tile_d1) of 2-byte elems, row stride
// stride_d0 elems, global -> LDS at lds_byte_off.  D# per ISA 08 §8.3/8.4.
// This toolchain exposes the 6-arg builtin (g0, g1, g2, g3, g4, cpol).
__device__ __forceinline__ void tdm_load_tile_2d(
    const void* gptr, unsigned lds_byte_off,
    unsigned tile_d0, unsigned tile_d1, unsigned stride_d0) {
    const unsigned long long ga = (unsigned long long)(size_t)gptr;
    u32x4 g0;
    g0[0] = 1u;                                                // count=1 (valid)
    g0[1] = lds_byte_off;                                      // lds_addr
    g0[2] = (unsigned)ga;                                      // global_addr lo
    g0[3] = (unsigned)((ga >> 32) & 0x01ffffffu) | (2u << 30); // hi | type=2
    i32x8 g1;
    g1[0] = (int)(1u << 16);          // data_size=1 -> 2 bytes/elem
    g1[1] = (int)(tile_d0 << 16);     // tensor_dim0[15:0]
    g1[2] = (int)(tile_d1 << 16);     // tensor_dim0 hi | tensor_dim1[15:0]
    g1[3] = (int)(tile_d0 << 16);     // tensor_dim1 hi | tile_dim0
    g1[4] = (int)tile_d1;             // tile_dim1 | tile_dim2=0
    g1[5] = (int)stride_d0;           // tensor_dim0_stride lo
    g1[6] = 0;
    g1[7] = 0;
    const i32x4 z4 = {0, 0, 0, 0};
    const i32x8 z8 = {0, 0, 0, 0, 0, 0, 0, 0};
    __builtin_amdgcn_tensor_load_to_lds(g0, g1, z4, z4, z8, 0);
}
#endif

// ---------------------------------------------------------------------------
// Pre-pack weights f32[K,N] -> WMMA B-fragment order, one u32 (k,k+1 bf16
// pair) per element: Bp[((nt*K/32 + kt)*32 + lane)*8 + j].
// ---------------------------------------------------------------------------
__global__ __launch_bounds__(256) void pack_b_kernel(
    const float* __restrict__ W, unsigned int* __restrict__ Bp,
    int N, int k32, int total) {
    const int i = blockIdx.x * 256 + threadIdx.x;
    if (i >= total) return;
    const int j    = i & 7;
    const int lane = (i >> 3) & 31;
    const int tile = i >> 8;
    const int kt   = tile % k32;
    const int nt   = tile / k32;
    const int half = lane >> 4, ln = lane & 15;
    const int n = nt * 16 + ln;
    const int k = kt * 32 + a_kpair(j, half);
    const unsigned lo = f2bf(W[(size_t)k * N + n]);
    const unsigned hi = f2bf(W[(size_t)(k + 1) * N + n]);
    Bp[i] = lo | (hi << 16);
}

// ---------------------------------------------------------------------------
// LayerNorm (scalar alpha/beta, unbiased std ddof=1, /(std+eps)) -> bf16.
// ---------------------------------------------------------------------------
__global__ __launch_bounds__(256) void ln_bf16_kernel(
    const float* __restrict__ x, const float* __restrict__ alpha,
    const float* __restrict__ beta, unsigned short* __restrict__ out) {
    __shared__ float rs[256], rq[256];
    __shared__ float mv[2];
    const int tid = threadIdx.x;
    const size_t row = blockIdx.x;
    float s = 0.f, q2 = 0.f;
    for (int i = tid; i < D_MODEL; i += 256) {
        float xv = x[row * D_MODEL + i];
        s += xv; q2 += xv * xv;
    }
    rs[tid] = s; rq[tid] = q2;
    __syncthreads();
    for (int st = 128; st > 0; st >>= 1) {
        if (tid < st) { rs[tid] += rs[tid + st]; rq[tid] += rq[tid + st]; }
        __syncthreads();
    }
    if (tid == 0) {
        float mean = rs[0] * (1.0f / D_MODEL);
        float var  = (rq[0] - (float)D_MODEL * mean * mean) * (1.0f / (D_MODEL - 1));
        mv[0] = mean;
        mv[1] = sqrtf(fmaxf(var, 0.0f));
    }
    __syncthreads();
    const float mean = mv[0], stdv = mv[1];
    const float a0 = alpha[0], b0 = beta[0];
    const float inv = a0 / (stdv + 1e-6f);
    for (int i = tid; i < D_MODEL; i += 256)
        out[row * D_MODEL + i] = f2bf((x[row * D_MODEL + i] - mean) * inv + b0);
}

// ---------------------------------------------------------------------------
// WMMA GEMM with pre-packed B: no LDS, no barriers.  256 thr = 8 waves,
// block tile 128(M) x 128(N); wave w owns rows m0+w*16, all 8 n-tiles.
// Per K-step/wave: 2 b128 A loads + 16 b128 B loads (L0-shared) + 8 WMMA.
// ---------------------------------------------------------------------------
template <int RELU, int OUT_BF16, int RES>
__global__ __launch_bounds__(256) void gemm_pk_kernel(
    const unsigned short* __restrict__ A, const unsigned int* __restrict__ Bp,
    const float* __restrict__ bias, const float* __restrict__ res,
    float* __restrict__ Cf, unsigned short* __restrict__ Cb,
    int M, int N, int K) {
    const int tid  = threadIdx.x;
    const int wave = tid >> 5, lane = tid & 31;
    const int half = lane >> 4, ln = lane & 15;
    const int m0 = blockIdx.y * 128;
    const int n0 = blockIdx.x * 128;
    const int k32 = K >> 5;
    const int row = m0 + wave * 16 + ln;

    v8f acc[8];
    for (int t = 0; t < 8; ++t)
        for (int r = 0; r < 8; ++r) acc[t][r] = 0.0f;

    const unsigned short* arow = A + (size_t)row * K;
    const int nt0 = n0 >> 4;

    for (int kt = 0; kt < k32; ++kt) {
        u32x8 au;   // A fragment: two b128 chunks (k = 8*half.. and 16+8*half..)
        {
            const uint4 alo = *(const uint4*)(arow + kt * 32 + half * 8);
            const uint4 ahi = *(const uint4*)(arow + kt * 32 + 16 + half * 8);
            au[0] = alo.x; au[1] = alo.y; au[2] = alo.z; au[3] = alo.w;
            au[4] = ahi.x; au[5] = ahi.y; au[6] = ahi.z; au[7] = ahi.w;
        }
        if (kt + 1 < k32)
            __builtin_prefetch(arow + (kt + 1) * 32 + half * 8, 0, 1);
#pragma unroll
        for (int t = 0; t < 8; ++t) {
            const unsigned int* bp =
                Bp + ((size_t)((nt0 + t) * k32 + kt) * 32 + lane) * 8;
            const uint4 b0 = *(const uint4*)bp;
            const uint4 b1 = *(const uint4*)(bp + 4);
            u32x8 bu;
            bu[0] = b0.x; bu[1] = b0.y; bu[2] = b0.z; bu[3] = b0.w;
            bu[4] = b1.x; bu[5] = b1.y; bu[6] = b1.z; bu[7] = b1.w;
            acc[t] = wmma_bf16(au, bu, acc[t]);
        }
    }

    // C layout: VGPR r -> row r + 8*half, col = lane&15
#pragma unroll
    for (int t = 0; t < 8; ++t) {
        const int ng = n0 + t * 16 + ln;
        const float bb = bias[ng];
#pragma unroll
        for (int r = 0; r < 8; ++r) {
            const int mg = m0 + wave * 16 + r + 8 * half;
            float v = acc[t][r] + bb;
            if (RES)  v += res[(size_t)mg * N + ng];
            if (RELU) v = fmaxf(v, 0.0f);
            if (OUT_BF16) Cb[(size_t)mg * N + ng] = f2bf(v);
            else          Cf[(size_t)mg * N + ng] = v;
        }
    }
}

// ---------------------------------------------------------------------------
// Flash attention: grid (S/64, H, B), 128 thr = 4 waves, 16 query rows/wave.
// Q fragments pre-scaled by 1/sqrt(DK) and register-resident.  K fragments
// direct from global (b128, L0-cached).  The 32x64 V tile is block-shared:
// staged into double-buffered LDS by the TDM (issue at top of iter i, wait at
// the bottom -> DMA overlaps a full iteration) with a manual fallback.
// P goes through a per-wave LDS bounce to reach A-fragment layout.
// ---------------------------------------------------------------------------
#define ATTN_SMEM_USHORTS (2 * 32 * 64 + 4 * 16 * 34)

__global__ __launch_bounds__(128) void attn_kernel(
    const unsigned short* __restrict__ Q, const unsigned short* __restrict__ Km,
    const unsigned short* __restrict__ Vm, const int* __restrict__ mask,
    unsigned short* __restrict__ ctx) {
    extern __shared__ unsigned short smem[];
    unsigned short* Vs = smem;                 // [2][32][64], TDM dest @ off 0/4096
    unsigned short* pl = smem + 2 * 32 * 64;   // [4][16][34]

    const int tid = threadIdx.x;
    const int wave = tid >> 5, lane = tid & 31, half = lane >> 4, ln = lane & 15;
    const int b = blockIdx.z, h = blockIdx.y;
    const int qbase = blockIdx.x * 64 + wave * 16;
    const unsigned short* vbase = Vm + (size_t)(b * SEQ) * D_MODEL + h * DK;

    // Q fragments (d 0..31 / 32..63), pre-scaled
    u32x8 aq0, aq1;
    {
        const unsigned short* qrow =
            Q + (size_t)(b * SEQ + qbase + ln) * D_MODEL + h * DK;
        const uint4 l0 = *(const uint4*)(qrow + half * 8);
        const uint4 l1 = *(const uint4*)(qrow + 16 + half * 8);
        const uint4 l2 = *(const uint4*)(qrow + 32 + half * 8);
        const uint4 l3 = *(const uint4*)(qrow + 48 + half * 8);
        aq0[0] = scale_pk(l0.x); aq0[1] = scale_pk(l0.y);
        aq0[2] = scale_pk(l0.z); aq0[3] = scale_pk(l0.w);
        aq0[4] = scale_pk(l1.x); aq0[5] = scale_pk(l1.y);
        aq0[6] = scale_pk(l1.z); aq0[7] = scale_pk(l1.w);
        aq1[0] = scale_pk(l2.x); aq1[1] = scale_pk(l2.y);
        aq1[2] = scale_pk(l2.z); aq1[3] = scale_pk(l2.w);
        aq1[4] = scale_pk(l3.x); aq1[5] = scale_pk(l3.y);
        aq1[6] = scale_pk(l3.z); aq1[7] = scale_pk(l3.w);
    }

    float m_r[8], l_r[8];
    v8f acc[4];
#pragma unroll
    for (int r = 0; r < 8; ++r) { m_r[r] = -1e30f; l_r[r] = 0.0f; }
    for (int t = 0; t < 4; ++t)
        for (int r = 0; r < 8; ++r) acc[t][r] = 0.0f;

    const int* mrow = mask + b * SEQ;

    // prologue: stage V tile for kv=0 into buffer 0
#if USE_TDM
    if (wave == 0) {
        tdm_load_tile_2d(vbase, 0u, DK, 32u, D_MODEL);
        __builtin_amdgcn_s_wait_tensorcnt(0);
    }
#else
    {
        const int r = tid >> 2, c = (tid & 3) * 16;
        const uint4 v0 = *(const uint4*)(vbase + (size_t)r * D_MODEL + c);
        const uint4 v1 = *(const uint4*)(vbase + (size_t)r * D_MODEL + c + 8);
        *(uint4*)&Vs[r * 64 + c]     = v0;
        *(uint4*)&Vs[r * 64 + c + 8] = v1;
    }
#endif
    __syncthreads();

    for (int kv = 0; kv < SEQ; kv += 32) {
        const int buf = (kv >> 5) & 1;
#if USE_TDM
        if (kv + 32 < SEQ && wave == 0)   // async: overlaps this iteration
            tdm_load_tile_2d(vbase + (size_t)(kv + 32) * D_MODEL,
                             (unsigned)((buf ^ 1) * 32 * 64 * 2), DK, 32u, D_MODEL);
#endif
        // ---- scores: two 16x16 tiles over keys kv..kv+31 (d=64 in 2 chunks)
        v8f c0, c1;
        for (int r = 0; r < 8; ++r) { c0[r] = 0.f; c1[r] = 0.f; }
        {
            const unsigned short* krow0 =
                Km + (size_t)(b * SEQ + kv + ln) * D_MODEL + h * DK + half * 8;
            const unsigned short* krow1 = krow0 + (size_t)16 * D_MODEL;
            c0 = wmma_bf16(aq0, load_frag_contig(krow0), c0);
            c0 = wmma_bf16(aq1, load_frag_contig(krow0 + 32), c0);
            c1 = wmma_bf16(aq0, load_frag_contig(krow1), c1);
            c1 = wmma_bf16(aq1, load_frag_contig(krow1 + 32), c1);
        }
        const bool msk0 = (mrow[kv + ln] == 0);
        const bool msk1 = (mrow[kv + 16 + ln] == 0);
        // ---- online softmax over the 32 new keys
        float p0[8], p1[8];
#pragma unroll
        for (int r = 0; r < 8; ++r) {
            float s0 = msk0 ? -1.0e9f : c0[r];
            float s1 = msk1 ? -1.0e9f : c1[r];
            float mr = fmaxf(s0, s1);
            mr = fmaxf(mr, __shfl_xor(mr, 1, 32));
            mr = fmaxf(mr, __shfl_xor(mr, 2, 32));
            mr = fmaxf(mr, __shfl_xor(mr, 4, 32));
            mr = fmaxf(mr, __shfl_xor(mr, 8, 32));
            const float mnew = fmaxf(m_r[r], mr);
            const float sc = __expf(m_r[r] - mnew);
            p0[r] = __expf(s0 - mnew);
            p1[r] = __expf(s1 - mnew);
            float rsum = p0[r] + p1[r];
            rsum += __shfl_xor(rsum, 1, 32);
            rsum += __shfl_xor(rsum, 2, 32);
            rsum += __shfl_xor(rsum, 4, 32);
            rsum += __shfl_xor(rsum, 8, 32);
            l_r[r] = l_r[r] * sc + rsum;
            m_r[r] = mnew;
#pragma unroll
            for (int t = 0; t < 4; ++t) acc[t][r] *= sc;
        }
        // ---- P bounce: C layout -> A-fragment layout (per-wave LDS, wave32
        // lockstep + in-order LDS pipe; wave_barrier blocks compiler reorder)
        __builtin_amdgcn_wave_barrier();
#pragma unroll
        for (int r = 0; r < 8; ++r) {
            const int prow = (wave * 16 + r + 8 * half) * 34;
            pl[prow + ln]      = f2bf(p0[r]);
            pl[prow + 16 + ln] = f2bf(p1[r]);
        }
        __builtin_amdgcn_wave_barrier();
        u32x8 pa;
#pragma unroll
        for (int j = 0; j < 8; ++j)
            pa[j] = *(const unsigned int*)&pl[(wave * 16 + ln) * 34 + a_kpair(j, half)];
        // ---- ctx += P * V  (V from the block-shared LDS tile)
#pragma unroll
        for (int t = 0; t < 4; ++t) {
            const int n = t * 16 + ln;
            u32x8 vb;
#pragma unroll
            for (int j = 0; j < 8; ++j) {
                const int k0 = a_kpair(j, half);
                vb[j] = (unsigned)Vs[(buf * 32 + k0) * 64 + n] |
                        ((unsigned)Vs[(buf * 32 + k0 + 1) * 64 + n] << 16);
            }
            acc[t] = wmma_bf16(pa, vb, acc[t]);
        }
        __syncthreads();   // all waves done reading Vs[buf]
        if (kv + 32 < SEQ) {
#if USE_TDM
            if (wave == 0) __builtin_amdgcn_s_wait_tensorcnt(0);
#else
            const int r = tid >> 2, c = (tid & 3) * 16;
            const unsigned short* vsrc = vbase + (size_t)(kv + 32 + r) * D_MODEL + c;
            const uint4 v0 = *(const uint4*)vsrc;
            const uint4 v1 = *(const uint4*)(vsrc + 8);
            *(uint4*)&Vs[((buf ^ 1) * 32 + r) * 64 + c]     = v0;
            *(uint4*)&Vs[((buf ^ 1) * 32 + r) * 64 + c + 8] = v1;
#endif
            __syncthreads();   // next buffer ready
        }
    }
    // ctx = acc / l, bf16 out for the WO GEMM
#pragma unroll
    for (int t = 0; t < 4; ++t) {
#pragma unroll
        for (int r = 0; r < 8; ++r) {
            const int srow = qbase + r + 8 * half;
            ctx[(size_t)(b * SEQ + srow) * D_MODEL + h * DK + t * 16 + ln] =
                f2bf(acc[t][r] / l_r[r]);
        }
    }
}

// ---------------------------------------------------------------------------
// Host-side orchestration
// ---------------------------------------------------------------------------
extern "C" void kernel_launch(void* const* d_in, const int* in_sizes, int n_in,
                              void* d_out, int out_size, void* d_ws, size_t ws_size,
                              hipStream_t stream) {
    (void)in_sizes; (void)n_in; (void)out_size; (void)ws_size;
    const float* x    = (const float*)d_in[0];
    const int*   mask = (const int*)d_in[1];
    const float* wq = (const float*)d_in[2];
    const float* bq = (const float*)d_in[3];
    const float* wk = (const float*)d_in[4];
    const float* bk = (const float*)d_in[5];
    const float* wv = (const float*)d_in[6];
    const float* bv = (const float*)d_in[7];
    const float* wo = (const float*)d_in[8];
    const float* bo = (const float*)d_in[9];
    const float* w1 = (const float*)d_in[10];
    const float* b1 = (const float*)d_in[11];
    const float* w2 = (const float*)d_in[12];
    const float* b2 = (const float*)d_in[13];
    const float* alpha1 = (const float*)d_in[14];
    const float* beta1  = (const float*)d_in[15];
    const float* alpha2 = (const float*)d_in[16];
    const float* beta2  = (const float*)d_in[17];
    float* out = (float*)d_out;

    char* wp = (char*)d_ws;
    auto alloc = [&](size_t bytes) -> void* {
        void* p = (void*)wp;
        wp += (bytes + 255) & ~(size_t)255;
        return p;
    };
    unsigned int*   wqp   = (unsigned int*)alloc((size_t)D_MODEL * D_MODEL * 2);
    unsigned int*   wkp   = (unsigned int*)alloc((size_t)D_MODEL * D_MODEL * 2);
    unsigned int*   wvp   = (unsigned int*)alloc((size_t)D_MODEL * D_MODEL * 2);
    unsigned int*   wop   = (unsigned int*)alloc((size_t)D_MODEL * D_MODEL * 2);
    unsigned int*   w1p   = (unsigned int*)alloc((size_t)D_MODEL * DFF * 2);
    unsigned int*   w2p   = (unsigned int*)alloc((size_t)DFF * D_MODEL * 2);
    unsigned short* xn16  = (unsigned short*)alloc((size_t)ROWS * D_MODEL * 2);
    unsigned short* q16   = (unsigned short*)alloc((size_t)ROWS * D_MODEL * 2);
    unsigned short* k16   = (unsigned short*)alloc((size_t)ROWS * D_MODEL * 2);
    unsigned short* v16   = (unsigned short*)alloc((size_t)ROWS * D_MODEL * 2);
    unsigned short* ctx16 = (unsigned short*)alloc((size_t)ROWS * D_MODEL * 2);
    float*          x1    = (float*)alloc((size_t)ROWS * D_MODEL * 4);
    unsigned short* h16   = (unsigned short*)alloc((size_t)ROWS * DFF * 2);

    // pre-pack weights into WMMA B-fragment order (fuses f32->bf16 convert)
    auto pack = [&](const float* W, unsigned int* Bp, int K, int N) {
        const int total = (K / 2) * N;
        pack_b_kernel<<<dim3((total + 255) / 256), 256, 0, stream>>>(
            W, Bp, N, K / 32, total);
    };
    pack(wq, wqp, D_MODEL, D_MODEL);
    pack(wk, wkp, D_MODEL, D_MODEL);
    pack(wv, wvp, D_MODEL, D_MODEL);
    pack(wo, wop, D_MODEL, D_MODEL);
    pack(w1, w1p, D_MODEL, DFF);
    pack(w2, w2p, DFF, D_MODEL);

    // LN1 -> bf16
    ln_bf16_kernel<<<dim3(ROWS), dim3(256), 0, stream>>>(x, alpha1, beta1, xn16);

    // Q/K/V projections (bf16 out)
    const dim3 gQKV(D_MODEL / 128, ROWS / 128);
    gemm_pk_kernel<0, 1, 0><<<gQKV, 256, 0, stream>>>(
        xn16, wqp, bq, nullptr, nullptr, q16, ROWS, D_MODEL, D_MODEL);
    gemm_pk_kernel<0, 1, 0><<<gQKV, 256, 0, stream>>>(
        xn16, wkp, bk, nullptr, nullptr, k16, ROWS, D_MODEL, D_MODEL);
    gemm_pk_kernel<0, 1, 0><<<gQKV, 256, 0, stream>>>(
        xn16, wvp, bv, nullptr, nullptr, v16, ROWS, D_MODEL, D_MODEL);

    // flash attention (dynamic LDS: V double-buffer + P bounce)
    attn_kernel<<<dim3(SEQ / 64, HEADS, BATCH), 128,
                  ATTN_SMEM_USHORTS * sizeof(unsigned short), stream>>>(
        q16, k16, v16, mask, ctx16);

    // x1 = x + ctx @ wo + bo  (f32 out, fused residual)
    gemm_pk_kernel<0, 0, 1><<<gQKV, 256, 0, stream>>>(
        ctx16, wop, bo, x, x1, nullptr, ROWS, D_MODEL, D_MODEL);

    // LN2 -> bf16 (reuse xn16)
    ln_bf16_kernel<<<dim3(ROWS), dim3(256), 0, stream>>>(x1, alpha2, beta2, xn16);

    // h = relu(xn2 @ w1 + b1)  (bf16 out)
    gemm_pk_kernel<1, 1, 0><<<dim3(DFF / 128, ROWS / 128), 256, 0, stream>>>(
        xn16, w1p, b1, nullptr, nullptr, h16, ROWS, DFF, D_MODEL);

    // out = x1 + h @ w2 + b2  (f32, fused residual -> d_out)
    gemm_pk_kernel<0, 0, 1><<<dim3(D_MODEL / 128, ROWS / 128), 256, 0, stream>>>(
        h16, w2p, b2, x1, out, nullptr, ROWS, D_MODEL, DFF);
}
